// GQAAbsorbedAttention_11012296146956
// MI455X (gfx1250) — matrix operand order repaired and checked by reading, not compiled, not run on hardware
//
#include <hip/hip_runtime.h>
#include <hip/hip_bf16.h>

// ---------------------------------------------------------------------------
// GQA attention for MI455X (gfx1250, wave32, WMMA f32_16x16x32_f16)
// ---------------------------------------------------------------------------

typedef __attribute__((ext_vector_type(16))) _Float16 v16h;
typedef __attribute__((ext_vector_type(8)))  float    v8f;

union FragU {
    v16h v;
    uint4 q[2];
};

__device__ __forceinline__ v8f wmma_f16(v16h a, v16h b, v8f c) {
    // 8 args: (neg_a, A, neg_b, B, c_mod, C, reuse_a, reuse_b)
    return __builtin_amdgcn_wmma_f32_16x16x32_f16(false, a, false, b, (short)0, c,
                                                  false, false);
}

// A fragment: 16 (M rows) x 32 (K), row-major, K contiguous, leading dim ld (halves).
// ISA layout: lane L -> row = L&15, khalf = L>>4 ;
//   VGPRs 0..3 <- K = khalf*8 + 0..7 ; VGPRs 4..7 <- K = 16 + khalf*8 + 0..7.
__device__ __forceinline__ v16h load_frag_a(const _Float16* p, int ld) {
    const int lane = threadIdx.x & 31;
    const int row  = lane & 15;
    const int kh   = lane >> 4;
    const _Float16* base = p + (size_t)row * (size_t)ld + kh * 8;
    FragU f;
    f.q[0] = *(const uint4*)(base);        // K = kh*8 .. kh*8+7
    f.q[1] = *(const uint4*)(base + 16);   // K = 16+kh*8 .. 16+kh*8+7
    return f.v;
}

// B fragment: 32 (K) x 16 (N cols). We hold B^T row-major (each output column is a
// K-contiguous row in memory, leading dim ld).
// ISA layout: lane L -> col = L&15, khalf = L>>4 ; K = khalf*8 + 0..7, then 16 + khalf*8 + 0..7.
__device__ __forceinline__ v16h load_frag_b(const _Float16* p, int ld) {
    const int lane = threadIdx.x & 31;
    const int col  = lane & 15;
    const int kh   = lane >> 4;
    const _Float16* base = p + (size_t)col * (size_t)ld + kh * 8;
    FragU f;
    f.q[0] = *(const uint4*)(base);        // K = kh*8 .. kh*8+7
    f.q[1] = *(const uint4*)(base + 16);   // K = 16+kh*8 .. 16+kh*8+7
    return f.v;
}

// ---------------------------------------------------------------------------
// fp32 -> f16 conversion
// ---------------------------------------------------------------------------
__global__ void cvt_f32_to_f16_kernel(const float* __restrict__ in,
                                      _Float16* __restrict__ out, int n) {
    int i = blockIdx.x * blockDim.x + threadIdx.x;
    if (i < n) out[i] = (_Float16)in[i];
}

// ---------------------------------------------------------------------------
// NT GEMM: C[M,N] = alpha * A[M,K] * B[N,K]^T, f16 inputs, f32 accumulate.
// Block = 256 threads = 8 waves (2 rows x 4 cols); wave tile 64x32 (4x2 WMMA
// tiles, 8 accumulators); block tile 128x128.
// Per k-step a wave issues 6 fragment loads for 8 WMMAs (~21 FLOP/byte).
// OMODE: 0 = f32 row-major out, 1 = f16 row-major out, 2 = f16 transposed out (C^T).
// M = gridDim.x*128, N = gridDim.y*128 (all shapes here are exact multiples).
// ---------------------------------------------------------------------------
template <int OMODE>
__global__ __launch_bounds__(256) void gemm_nt_kernel(
    const _Float16* __restrict__ A, const _Float16* __restrict__ B,
    void* __restrict__ Cvoid, int K, int lda, int ldb, int ldc, float alpha) {

    const int tid  = threadIdx.x;
    const int lane = tid & 31;
    const int wib  = tid >> 5;                         // wave in block: 0..7
    const int mbase = blockIdx.x * 128 + (wib >> 2) * 64;   // 2 wave-rows
    const int nbase = blockIdx.y * 128 + (wib & 3) * 32;    // 4 wave-cols

    const v8f vzero = {0.f, 0.f, 0.f, 0.f, 0.f, 0.f, 0.f, 0.f};
    v8f acc[4][2];
#pragma unroll
    for (int i = 0; i < 4; i++)
#pragma unroll
        for (int j = 0; j < 2; j++) acc[i][j] = vzero;

    const _Float16* Ar[4];
#pragma unroll
    for (int i = 0; i < 4; i++) Ar[i] = A + (size_t)(mbase + i * 16) * (size_t)lda;
    const _Float16* Bc[2];
#pragma unroll
    for (int j = 0; j < 2; j++) Bc[j] = B + (size_t)(nbase + j * 16) * (size_t)ldb;

    for (int k = 0; k < K; k += 32) {
        // prefetch ahead into L1/L2 (speculative; OOB prefetch is dropped)
        __builtin_prefetch(Ar[0] + k + 256, 0, 0);
        __builtin_prefetch(Bc[0] + k + 256, 0, 0);

        v16h af[4], bf[2];
#pragma unroll
        for (int i = 0; i < 4; i++) af[i] = load_frag_a(Ar[i] + k, lda);
#pragma unroll
        for (int j = 0; j < 2; j++) bf[j] = load_frag_b(Bc[j] + k, ldb);
#pragma unroll
        for (int i = 0; i < 4; i++)
#pragma unroll
            for (int j = 0; j < 2; j++) acc[i][j] = wmma_f16(af[i], bf[j], acc[i][j]);
    }

    // C tile layout: lane -> col = lane&15, rows = (lane>>4)*8 + r  (r = vgpr 0..7)
    const int cm = (lane >> 4) * 8;
    const int cn = lane & 15;

    if (OMODE == 0) {
        float* C = (float*)Cvoid;
#pragma unroll
        for (int i = 0; i < 4; i++)
#pragma unroll
            for (int j = 0; j < 2; j++) {
                const size_t rowb = (size_t)(mbase + i * 16 + cm);
                const int col = nbase + j * 16 + cn;
#pragma unroll
                for (int r = 0; r < 8; r++)
                    C[(rowb + r) * (size_t)ldc + col] = acc[i][j][r] * alpha;
            }
    } else if (OMODE == 1) {
        _Float16* C = (_Float16*)Cvoid;
#pragma unroll
        for (int i = 0; i < 4; i++)
#pragma unroll
            for (int j = 0; j < 2; j++) {
                const size_t rowb = (size_t)(mbase + i * 16 + cm);
                const int col = nbase + j * 16 + cn;
#pragma unroll
                for (int r = 0; r < 8; r++)
                    C[(rowb + r) * (size_t)ldc + col] = (_Float16)(acc[i][j][r] * alpha);
            }
    } else {
        // Transposed f16 store: C^T[n, m], rows m contiguous per lane -> b128 stores.
        _Float16* C = (_Float16*)Cvoid;
#pragma unroll
        for (int i = 0; i < 4; i++)
#pragma unroll
            for (int j = 0; j < 2; j++) {
                union { uint4 q; _Float16 h[8]; } u;
#pragma unroll
                for (int r = 0; r < 8; r++) u.h[r] = (_Float16)(acc[i][j][r] * alpha);
                *(uint4*)&C[(size_t)(nbase + j * 16 + cn) * (size_t)ldc +
                            (mbase + i * 16 + cm)] = u.q;
            }
    }
}

// ---------------------------------------------------------------------------
// Flash attention: one wave per (b, head, 16-query tile). T=2048, head_dim=128.
// Q16 : [B*T, 2048] f16, pre-scaled by 1/sqrt(128)
// K16 : [B*T, 1024] f16 (kv features)
// Vt16: [1024, B*T] f16 (V transposed: feature-major, t contiguous)
// Od  : [B*T, 2048] f16 attention output (t-major, feature contiguous)
// ---------------------------------------------------------------------------
#define T_SEQ 2048

__global__ __launch_bounds__(256) void attn_fa_kernel(
    const _Float16* __restrict__ Q16, const _Float16* __restrict__ K16,
    const _Float16* __restrict__ Vt16, _Float16* __restrict__ Od) {

    __shared__ __align__(16) _Float16 smem[8][16 * 32];  // per-wave P staging tile

    const int tid  = threadIdx.x;
    const int lane = tid & 31;
    const int wib  = tid >> 5;
    const int w    = blockIdx.x * 8 + wib;    // 0 .. 4095
    const int mt   = w & 127;                 // query tile (T/16 = 128)
    const int h    = (w >> 7) & 15;           // q head
    const int b    = w >> 11;                 // batch
    const int kv   = h >> 1;                  // kv head (G = 2)

    const int ldq = 2048, ldk = 1024, ldvt = 4096, ldo = 2048;

    // Preload Q fragments for this 16-row tile (4 k-steps over head_dim=128).
    const _Float16* Qp = Q16 + (size_t)(b * T_SEQ + mt * 16) * ldq + h * 128;
    v16h qf0 = load_frag_a(Qp + 0,  ldq);
    v16h qf1 = load_frag_a(Qp + 32, ldq);
    v16h qf2 = load_frag_a(Qp + 64, ldq);
    v16h qf3 = load_frag_a(Qp + 96, ldq);

    const _Float16* Kb = K16 + (size_t)(b * T_SEQ) * ldk + kv * 128;
    const _Float16* Vb = Vt16 + (size_t)(kv * 128) * ldvt + (size_t)(b * T_SEQ);

    const v8f vzero = {0.f, 0.f, 0.f, 0.f, 0.f, 0.f, 0.f, 0.f};
    v8f o[8];
#pragma unroll
    for (int t = 0; t < 8; t++) o[t] = vzero;

    float m_r[8], l_r[8];
#pragma unroll
    for (int r = 0; r < 8; r++) { m_r[r] = -1e30f; l_r[r] = 0.f; }

    _Float16* pb = smem[wib];
    const int prow = (lane >> 4) * 8;   // C-layout: first row this lane holds
    const int pcol = lane & 15;         // C-layout: column this lane holds

    for (int s0 = 0; s0 < T_SEQ; s0 += 32) {
        // ---- S = Q * K^T for 32 keys (two 16x16 C tiles) ----
        v8f sc0 = vzero, sc1 = vzero;
        {
            const _Float16* k0 = Kb + (size_t)(s0)      * ldk;
            const _Float16* k1 = Kb + (size_t)(s0 + 16) * ldk;
            sc0 = wmma_f16(qf0, load_frag_b(k0 + 0,  ldk), sc0);
            sc0 = wmma_f16(qf1, load_frag_b(k0 + 32, ldk), sc0);
            sc0 = wmma_f16(qf2, load_frag_b(k0 + 64, ldk), sc0);
            sc0 = wmma_f16(qf3, load_frag_b(k0 + 96, ldk), sc0);
            sc1 = wmma_f16(qf0, load_frag_b(k1 + 0,  ldk), sc1);
            sc1 = wmma_f16(qf1, load_frag_b(k1 + 32, ldk), sc1);
            sc1 = wmma_f16(qf2, load_frag_b(k1 + 64, ldk), sc1);
            sc1 = wmma_f16(qf3, load_frag_b(k1 + 96, ldk), sc1);
        }

        // ---- online softmax over these 32 keys ----
        float corr[8];
#pragma unroll
        for (int r = 0; r < 8; r++) {
            float mx = fmaxf(sc0[r], sc1[r]);
            mx = fmaxf(mx, __shfl_xor(mx, 1, 32));
            mx = fmaxf(mx, __shfl_xor(mx, 2, 32));
            mx = fmaxf(mx, __shfl_xor(mx, 4, 32));
            mx = fmaxf(mx, __shfl_xor(mx, 8, 32));   // row-wide max (16 lanes/half)
            const float nm = fmaxf(m_r[r], mx);
            corr[r] = __expf(m_r[r] - nm);
            m_r[r]  = nm;
            const float p0 = __expf(sc0[r] - nm);
            const float p1 = __expf(sc1[r] - nm);
            sc0[r] = p0;
            sc1[r] = p1;
            float rs = p0 + p1;
            rs += __shfl_xor(rs, 1, 32);
            rs += __shfl_xor(rs, 2, 32);
            rs += __shfl_xor(rs, 4, 32);
            rs += __shfl_xor(rs, 8, 32);
            l_r[r] = l_r[r] * corr[r] + rs;
        }
#pragma unroll
        for (int t = 0; t < 8; t++) {
#pragma unroll
            for (int r = 0; r < 8; r++) o[t][r] *= corr[r];
        }

        // ---- P (C layout) -> LDS row-major 16x32 f16 ----
#pragma unroll
        for (int r = 0; r < 8; r++) {
            pb[(prow + r) * 32 + pcol]      = (_Float16)sc0[r];
            pb[(prow + r) * 32 + 16 + pcol] = (_Float16)sc1[r];
        }
        // Per-wave buffer: same-wave LDS ops are in-order; compiler inserts dscnt wait.

        // ---- P A-fragment from LDS, then O += P * V (8 d-tiles) ----
        v16h pf = load_frag_a(pb, 32);
#pragma unroll
        for (int t = 0; t < 8; t++) {
            v16h vf = load_frag_b(Vb + (size_t)(t * 16) * ldvt + s0, ldvt);
            o[t] = wmma_f16(pf, vf, o[t]);
        }
    }

    // ---- epilogue: O /= l, store f16 attention output ----
    float inv[8];
#pragma unroll
    for (int r = 0; r < 8; r++) inv[r] = 1.0f / l_r[r];

    const size_t orow0 = (size_t)(b * T_SEQ + mt * 16);
#pragma unroll
    for (int t = 0; t < 8; t++) {
#pragma unroll
        for (int r = 0; r < 8; r++) {
            Od[(orow0 + prow + r) * (size_t)ldo + h * 128 + t * 16 + pcol] =
                (_Float16)(o[t][r] * inv[r]);
        }
    }
}

// ---------------------------------------------------------------------------
// Launch
// ---------------------------------------------------------------------------
extern "C" void kernel_launch(void* const* d_in, const int* in_sizes, int n_in,
                              void* d_out, int out_size, void* d_ws, size_t ws_size,
                              hipStream_t stream) {
    (void)in_sizes; (void)n_in; (void)out_size;

    constexpr int Bc = 2, Tn = 2048, Dm = 2048;
    constexpr int NH = 16, NKV = 8, HD = 128;
    constexpr int BT = Bc * Tn;                 // 4096
    constexpr int QD = NH * HD;                 // 2048
    constexpr int KD = NKV * HD;                // 1024

    const float* x  = (const float*)d_in[0];
    const float* Wq = (const float*)d_in[1];
    const float* Wk = (const float*)d_in[2];
    const float* Wv = (const float*)d_in[3];
    const float* Wo = (const float*)d_in[4];
    float* out = (float*)d_out;

    char* ws = (char*)d_ws;
    size_t off = 0;
    auto alloc = [&](size_t nhalf) -> _Float16* {
        _Float16* p = (_Float16*)(ws + off);
        off += (nhalf * sizeof(_Float16) + 255) & ~(size_t)255;
        return p;
    };
    _Float16* x16  = alloc((size_t)BT * Dm);
    _Float16* Wq16 = alloc((size_t)QD * Dm);
    _Float16* Wk16 = alloc((size_t)KD * Dm);
    _Float16* Wv16 = alloc((size_t)KD * Dm);
    _Float16* Wo16 = alloc((size_t)Dm * QD);
    _Float16* Q16  = alloc((size_t)BT * QD);
    _Float16* K16  = alloc((size_t)BT * KD);
    _Float16* Vt16 = alloc((size_t)KD * BT);    // V transposed: [feature, b*T + t]
    _Float16* A16  = alloc((size_t)BT * QD);    // attention output, f16
    if (off > ws_size) return;                  // workspace too small: nothing safe to do

    auto cvt = [&](const float* src, _Float16* dst, int n) {
        cvt_f32_to_f16_kernel<<<(n + 255) / 256, 256, 0, stream>>>(src, dst, n);
    };
    cvt(x,  x16,  BT * Dm);
    cvt(Wq, Wq16, QD * Dm);
    cvt(Wk, Wk16, KD * Dm);
    cvt(Wv, Wv16, KD * Dm);
    cvt(Wo, Wo16, Dm * QD);

    const float qscale = 0.08838834764831845f;  // 1/sqrt(128)

    // Q = (x @ Wq^T) * qscale   -> f16 row-major [BT, 2048]
    gemm_nt_kernel<1><<<dim3(BT / 128, QD / 128), 256, 0, stream>>>(
        x16, Wq16, Q16, Dm, Dm, Dm, QD, qscale);
    // K = x @ Wk^T              -> f16 row-major [BT, 1024]
    gemm_nt_kernel<1><<<dim3(BT / 128, KD / 128), 256, 0, stream>>>(
        x16, Wk16, K16, Dm, Dm, Dm, KD, 1.0f);
    // V^T = (x @ Wv^T)^T        -> f16 [1024, BT] (t contiguous)
    gemm_nt_kernel<2><<<dim3(BT / 128, KD / 128), 256, 0, stream>>>(
        x16, Wv16, Vt16, Dm, Dm, Dm, BT, 1.0f);

    // Flash attention: 4096 waves = 512 blocks * 8 waves
    attn_fa_kernel<<<dim3((Bc * NH * (Tn / 16)) / 8), 256, 0, stream>>>(
        Q16, K16, Vt16, A16);

    // out = attn @ Wo^T  -> f32 [BT, 2048]
    gemm_nt_kernel<0><<<dim3(BT / 128, Dm / 128), 256, 0, stream>>>(
        A16, Wo16, out, QD, QD, QD, Dm, 1.0f);
}
